// TransformerBlock_48567490183872
// MI455X (gfx1250) — matrix operand order repaired and checked by reading, not compiled, not run on hardware
//
#include <hip/hip_runtime.h>
#include <hip/hip_bf16.h>

// ---------------------------------------------------------------------------
// Transformer block for gfx1250 (MI455X).  All six GEMMs (QKV, O-proj,
// FFN x2, ~550 GFLOP total) run through v_wmma_f32_16x16x32_f16 with fp32
// accumulation.  Tile staging uses the CDNA5 async global->LDS DMA path
// (global_load_async_to_lds_b128, ASYNCcnt) with a 3-buffer / depth-2
// software pipeline: the DMA for tile t+2 overlaps two iterations of WMMA,
// drained with partial s_wait_asynccnt (in-order completion).  Softmax /
// layernorm / rank-1 attention glue are fp32 VALU.
// ---------------------------------------------------------------------------

typedef __attribute__((ext_vector_type(16))) _Float16 v16h;
typedef __attribute__((ext_vector_type(8)))  float    v8f;

namespace {
constexpr int Sq   = 8192;
constexpr int Hh   = 16;
constexpr int Dd   = 64;
constexpr int Ed   = 1024;
constexpr int FFd  = 2048;
constexpr int Nbat = 4;
constexpr int Mtot = Nbat * Sq;     // 32768 tokens

constexpr int BM = 128, BN = 128, BK = 32;
constexpr int LDSS = 40;            // padded LDS row stride (halves); 80B, 16B-aligned
constexpr int NBUF = 3;             // async pipeline buffers (3 x 20KB = 60KB LDS)
}

// ---- fp32 -> f16 elementwise --------------------------------------------------
__global__ __launch_bounds__(256) void cvt_f16_kernel(const float* __restrict__ src,
                                                      _Float16* __restrict__ dst, int n) {
  int i = blockIdx.x * 256 + threadIdx.x;
  if (i < n) dst[i] = (_Float16)src[i];
}

// ---- fp32 (Kd x Nd) -> f16 transposed (Nd x Kd) -------------------------------
__global__ __launch_bounds__(256) void transpose_f16_kernel(const float* __restrict__ W,
                                                            _Float16* __restrict__ Wt,
                                                            int Kd, int Nd) {
  int i = blockIdx.x * 256 + threadIdx.x;
  if (i >= Kd * Nd) return;
  int k = i / Nd, n = i - k * Nd;
  Wt[(size_t)n * Kd + k] = (_Float16)W[i];
}

// ---- WMMA GEMM: C[MxN] = A[MxK](f16,row) * Bt[NxK](f16,row) (+bias)(+relu) ----
__global__ __launch_bounds__(256)
void gemm_f16_kernel(const _Float16* __restrict__ A, const _Float16* __restrict__ Bt,
                     const float* __restrict__ bias, float* __restrict__ outF,
                     _Float16* __restrict__ outH, int M, int N, int K, int relu) {
  __shared__ __align__(16) _Float16 As[NBUF][BM * LDSS];
  __shared__ __align__(16) _Float16 Bs[NBUF][BN * LDSS];

  const int tid  = threadIdx.x;
  const int wave = tid >> 5;
  const int lane = tid & 31;
  const int half = lane >> 4;     // fragment half (ISA 16-bit layouts)
  const int mrow = lane & 15;

  const int bm = blockIdx.y * BM;
  const int bn = blockIdx.x * BN;
  const int wm = (wave & 3) * 32; // 4 waves along M
  const int wn = (wave >> 2) * 64;// 2 waves along N

  const _Float16* Aptr = A  + (size_t)bm * K;
  const _Float16* Bptr = Bt + (size_t)bn * K;

  // per-thread tile-staging coordinates: 2 chunks of 8 halves for A and for B
  const int srow0 = tid >> 2;                 // rows 0..63
  const int srow1 = (tid + 256) >> 2;         // rows 64..127
  const int sch   = (tid & 3) * 8;            // half-element chunk within a 32-wide row

  // async DMA of one 128x32 A tile + 128x32 Bt tile into LDS buffer `buf`
  auto stage = [&](int buf, int kt) {
    uint32_t la0 = (uint32_t)(uintptr_t)(&As[buf][srow0 * LDSS + sch]);
    uint32_t la1 = (uint32_t)(uintptr_t)(&As[buf][srow1 * LDSS + sch]);
    uint32_t lb0 = (uint32_t)(uintptr_t)(&Bs[buf][srow0 * LDSS + sch]);
    uint32_t lb1 = (uint32_t)(uintptr_t)(&Bs[buf][srow1 * LDSS + sch]);
    uint64_t ga0 = (uint64_t)(uintptr_t)(Aptr + (size_t)srow0 * K + kt + sch);
    uint64_t ga1 = (uint64_t)(uintptr_t)(Aptr + (size_t)srow1 * K + kt + sch);
    uint64_t gb0 = (uint64_t)(uintptr_t)(Bptr + (size_t)srow0 * K + kt + sch);
    uint64_t gb1 = (uint64_t)(uintptr_t)(Bptr + (size_t)srow1 * K + kt + sch);
    asm volatile("global_load_async_to_lds_b128 %0, %1, off" :: "v"(la0), "v"(ga0) : "memory");
    asm volatile("global_load_async_to_lds_b128 %0, %1, off" :: "v"(la1), "v"(ga1) : "memory");
    asm volatile("global_load_async_to_lds_b128 %0, %1, off" :: "v"(lb0), "v"(gb0) : "memory");
    asm volatile("global_load_async_to_lds_b128 %0, %1, off" :: "v"(lb1), "v"(gb1) : "memory");
  };

  v8f c[2][4] = {};

  const int nk = K / BK;
  // pipeline prologue: tiles 0 and 1 in flight; wait for tile 0 only (in-order)
  stage(0, 0);
  if (nk > 1) {
    stage(1, BK);
    asm volatile("s_wait_asynccnt 0x4" ::: "memory");
  } else {
    asm volatile("s_wait_asynccnt 0x0" ::: "memory");
  }
  __syncthreads();

  int rd = 0;                                  // rd == t % NBUF
  for (int t = 0; t < nk; ++t) {
    if (t + 2 < nk) {                          // DMA two tiles ahead
      int wr = rd + 2; if (wr >= NBUF) wr -= NBUF;
      stage(wr, (t + 2) * BK);
    }
    if (t + 4 < nk) {                          // GL2 warm-up beyond async window
      __builtin_prefetch(Aptr + (size_t)srow0 * K + (t + 4) * BK + sch, 0, 1);
      __builtin_prefetch(Bptr + (size_t)srow0 * K + (t + 4) * BK + sch, 0, 1);
    }

    const _Float16* Abase = &As[rd][0];
    const _Float16* Bbase = &Bs[rd][0];
    // A fragment 16x32: lanes 0-15 -> K 0..7 & 16..23, lanes 16-31 -> K 8..15 & 24..31
    v16h a[2], b[4];
    #pragma unroll
    for (int i = 0; i < 2; ++i) {
      const _Float16* p = Abase + (wm + i * 16 + mrow) * LDSS + half * 8;
      ((uint4*)&a[i])[0] = *(const uint4*)(p);
      ((uint4*)&a[i])[1] = *(const uint4*)(p + 16);
    }
    // B fragment 32x16 (from Bt rows): lanes 0-15 -> K 0..7 & 16..23, lanes 16-31 -> K 8..15 & 24..31
    #pragma unroll
    for (int j = 0; j < 4; ++j) {
      const _Float16* p = Bbase + (wn + j * 16 + mrow) * LDSS + half * 8;
      ((uint4*)&b[j])[0] = *(const uint4*)(p);
      ((uint4*)&b[j])[1] = *(const uint4*)(p + 16);
    }
    #pragma unroll
    for (int i = 0; i < 2; ++i)
      #pragma unroll
      for (int j = 0; j < 4; ++j)
        c[i][j] = __builtin_amdgcn_wmma_f32_16x16x32_f16(
            false, a[i], false, b[j], (short)0, c[i][j], false, false);

    // Drain my LDS reads; partially drain ASYNCcnt so tile t+1 is resident
    // (in-order completion: <=4 outstanding leaves only stage(t+2) in flight).
    asm volatile("s_wait_dscnt 0x0" ::: "memory");
    if (t + 2 < nk)      asm volatile("s_wait_asynccnt 0x4" ::: "memory");
    else                 asm volatile("s_wait_asynccnt 0x0" ::: "memory");
    __syncthreads();

    ++rd; if (rd == NBUF) rd = 0;
  }

  // C/D layout: lane -> n = lane%16, vgpr v -> m = v + 8*(lane/16)
  #pragma unroll
  for (int i = 0; i < 2; ++i) {
    #pragma unroll
    for (int j = 0; j < 4; ++j) {
      int n = bn + wn + j * 16 + mrow;
      float bv = bias ? bias[n] : 0.f;
      #pragma unroll
      for (int v = 0; v < 8; ++v) {
        int m = bm + wm + i * 16 + v + half * 8;
        float val = c[i][j][v] + bv;
        if (relu) val = fmaxf(val, 0.f);
        size_t o = (size_t)m * N + n;
        if (outF) outF[o] = val;
        if (outH) outH[o] = (_Float16)val;
      }
    }
  }
}

// ---- energy[n,q,k,s] = (Q_row(q) . K_row(k)) / 32 -----------------------------
__global__ __launch_bounds__(256)
void energy_kernel(const float* __restrict__ Q, const float* __restrict__ K,
                   float* __restrict__ E) {
  int idx = blockIdx.x * 256 + threadIdx.x;       // (n*Sq+s)*16 + q
  if (idx >= Mtot * Hh) return;
  int q  = idx & 15;
  int ns = idx >> 4;
  int n  = ns >> 13;                              // / 8192
  int s  = ns & (Sq - 1);
  const float* qp = Q + (size_t)ns * Ed + q * Dd;
  const float* kp = K + (size_t)ns * Ed;
  float qr[Dd];
  #pragma unroll
  for (int d = 0; d < Dd; ++d) qr[d] = qp[d];
  #pragma unroll
  for (int k = 0; k < Hh; ++k) {
    float acc = 0.f;
    #pragma unroll
    for (int d = 0; d < Dd; ++d) acc += qr[d] * kp[k * Dd + d];
    E[(((size_t)(n * Hh + q) * Hh + k) * Sq) + s] = acc * 0.03125f;  // 1/sqrt(1024)
  }
}

// ---- in-place softmax over s (rows of length 8192) ----------------------------
__global__ __launch_bounds__(256) void softmax_kernel(float* __restrict__ att) {
  __shared__ float rmax[8], rsum[8];
  float* row = att + (size_t)blockIdx.x * Sq;
  const int t = threadIdx.x;
  float v[32];
  float mx = -3.0e38f;
  #pragma unroll
  for (int i = 0; i < 32; ++i) { v[i] = row[t + i * 256]; mx = fmaxf(mx, v[i]); }
  for (int o = 16; o > 0; o >>= 1) mx = fmaxf(mx, __shfl_xor(mx, o, 32));
  if ((t & 31) == 0) rmax[t >> 5] = mx;
  __syncthreads();
  float m = rmax[0];
  #pragma unroll
  for (int i = 1; i < 8; ++i) m = fmaxf(m, rmax[i]);
  float s = 0.f;
  #pragma unroll
  for (int i = 0; i < 32; ++i) { v[i] = __expf(v[i] - m); s += v[i]; }
  for (int o = 16; o > 0; o >>= 1) s += __shfl_xor(s, o, 32);
  if ((t & 31) == 0) rsum[t >> 5] = s;
  __syncthreads();
  float tot = 0.f;
  #pragma unroll
  for (int i = 0; i < 8; ++i) tot += rsum[i];
  float inv = 1.f / tot;
  #pragma unroll
  for (int i = 0; i < 32; ++i) row[t + i * 256] = v[i] * inv;
}

// ---- A[n,q,s] = sum_k att[n,q,k,s] -------------------------------------------
__global__ __launch_bounds__(256)
void asum_kernel(const float* __restrict__ att, float* __restrict__ A) {
  int idx = blockIdx.x * 256 + threadIdx.x;       // (n*Hh+q)*Sq + s
  if (idx >= Nbat * Hh * Sq) return;
  int s  = idx & (Sq - 1);
  int nq = idx >> 13;
  const float* p = att + ((size_t)nq * Hh) * Sq + s;
  float acc = 0.f;
  #pragma unroll
  for (int k = 0; k < Hh; ++k) acc += p[(size_t)k * Sq];
  A[idx] = acc;
}

// ---- Vs[n,s,d] = sum_h V[n*Sq+s, h*64+d] -------------------------------------
__global__ __launch_bounds__(256)
void vsum_kernel(const float* __restrict__ V, float* __restrict__ Vs) {
  int idx = blockIdx.x * 256 + threadIdx.x;       // ns*64 + d
  if (idx >= Mtot * Dd) return;
  int d  = idx & (Dd - 1);
  int ns = idx >> 6;
  const float* p = V + (size_t)ns * Ed + d;
  float acc = 0.f;
  #pragma unroll
  for (int h = 0; h < Hh; ++h) acc += p[h * Dd];
  Vs[idx] = acc;
}

// ---- out2 (reshape-bug layout, f16): row = n*8192+q*512+s/16, col=(s%16)*64+d -
__global__ __launch_bounds__(256)
void out2_kernel(const float* __restrict__ A, const float* __restrict__ Vs,
                 _Float16* __restrict__ out2) {
  int idx = blockIdx.x * 256 + threadIdx.x;       // (n*Hh+q)*Sq + s
  if (idx >= Nbat * Hh * Sq) return;
  int s  = idx & (Sq - 1);
  int nq = idx >> 13;
  int n  = nq >> 4, q = nq & 15;
  float a = A[idx];
  const float* vp = Vs + ((size_t)n * Sq + s) * Dd;
  _Float16* op = out2 + ((size_t)(n * Sq + q * 512 + (s >> 4)) * Ed) + (s & 15) * Dd;
  #pragma unroll
  for (int d = 0; d < Dd; ++d) op[d] = (_Float16)(a * vp[d]);
}

// ---- layernorm(a+b) * g + beta; fp32 out + optional f16 out -------------------
__global__ __launch_bounds__(256)
void ln_kernel(const float* __restrict__ a, const float* __restrict__ b,
               const float* __restrict__ g, const float* __restrict__ beta,
               float* __restrict__ outF, _Float16* __restrict__ outH) {
  __shared__ float rs[8], rs2[8];
  const int row = blockIdx.x, t = threadIdx.x;
  const float* pa = a + (size_t)row * Ed;
  const float* pb = b + (size_t)row * Ed;
  float v[4], s = 0.f, s2 = 0.f;
  #pragma unroll
  for (int i = 0; i < 4; ++i) {
    float x = pa[t + 256 * i] + pb[t + 256 * i];
    v[i] = x; s += x; s2 += x * x;
  }
  for (int o = 16; o > 0; o >>= 1) { s += __shfl_xor(s, o, 32); s2 += __shfl_xor(s2, o, 32); }
  if ((t & 31) == 0) { rs[t >> 5] = s; rs2[t >> 5] = s2; }
  __syncthreads();
  s = 0.f; s2 = 0.f;
  #pragma unroll
  for (int i = 0; i < 8; ++i) { s += rs[i]; s2 += rs2[i]; }
  float mean = s * (1.f / Ed);
  float var  = s2 * (1.f / Ed) - mean * mean;
  float inv  = rsqrtf(var + 1e-5f);
  #pragma unroll
  for (int i = 0; i < 4; ++i) {
    int c = t + 256 * i;
    float y = (v[i] - mean) * inv * g[c] + beta[c];
    if (outF) outF[(size_t)row * Ed + c] = y;
    if (outH) outH[(size_t)row * Ed + c] = (_Float16)y;
  }
}

// ---------------------------------------------------------------------------
extern "C" void kernel_launch(void* const* d_in, const int* in_sizes, int n_in,
                              void* d_out, int out_size, void* d_ws, size_t ws_size,
                              hipStream_t stream) {
  const float* x   = (const float*)d_in[0];
  const float* wq  = (const float*)d_in[1];
  const float* wk  = (const float*)d_in[2];
  const float* wv  = (const float*)d_in[3];
  const float* wo  = (const float*)d_in[4];
  const float* bo  = (const float*)d_in[5];
  const float* w1  = (const float*)d_in[6];
  const float* b1  = (const float*)d_in[7];
  const float* w2  = (const float*)d_in[8];
  const float* b2  = (const float*)d_in[9];
  const float* g1  = (const float*)d_in[10];
  const float* be1 = (const float*)d_in[11];
  const float* g2  = (const float*)d_in[12];
  const float* be2 = (const float*)d_in[13];

  char* ws = (char*)d_ws;
  const size_t MB = 1024ull * 1024ull;
  // time-multiplexed workspace (~656 MB peak)
  float*    Qf    = (float*)(ws + 0 * MB);      // phase 1-2
  float*    Kf    = (float*)(ws + 128 * MB);
  float*    Vf    = (float*)(ws + 256 * MB);
  float*    Of    = (float*)(ws + 0 * MB);      // phase 3 (Q dead)
  float*    x1    = (float*)(ws + 128 * MB);    // phase 3-4 (K dead)
  float*    ff    = (float*)(ws + 256 * MB);    // phase 4 (V dead)
  _Float16* xb    = (_Float16*)(ws + 384 * MB); // phase 1
  _Float16* x1b   = (_Float16*)(ws + 384 * MB); // phase 3-4 (xb dead)
  _Float16* out2b = (_Float16*)(ws + 448 * MB);
  float*    att   = (float*)(ws + 512 * MB);    // phase 2 (32 MB)
  _Float16* h1b   = (_Float16*)(ws + 512 * MB); // phase 4 (att dead)
  float*    Asum  = (float*)(ws + 544 * MB);    // phase 2
  float*    Vs    = (float*)(ws + 546 * MB);    // phase 2
  _Float16* wqt   = (_Float16*)(ws + 640 * MB);
  _Float16* wkt   = (_Float16*)(ws + 642 * MB);
  _Float16* wvt   = (_Float16*)(ws + 644 * MB);
  _Float16* wot   = (_Float16*)(ws + 646 * MB);
  _Float16* w1t   = (_Float16*)(ws + 648 * MB);
  _Float16* w2t   = (_Float16*)(ws + 652 * MB);

  // ---- precision conversion (fp32 -> f16; weights pre-transposed to NxK) ----
  cvt_f16_kernel<<<(Mtot * Ed + 255) / 256, 256, 0, stream>>>(x, xb, Mtot * Ed);
  transpose_f16_kernel<<<(Ed * Ed + 255) / 256, 256, 0, stream>>>(wq, wqt, Ed, Ed);
  transpose_f16_kernel<<<(Ed * Ed + 255) / 256, 256, 0, stream>>>(wk, wkt, Ed, Ed);
  transpose_f16_kernel<<<(Ed * Ed + 255) / 256, 256, 0, stream>>>(wv, wvt, Ed, Ed);
  transpose_f16_kernel<<<(Ed * Ed + 255) / 256, 256, 0, stream>>>(wo, wot, Ed, Ed);
  transpose_f16_kernel<<<(Ed * FFd + 255) / 256, 256, 0, stream>>>(w1, w1t, Ed, FFd);
  transpose_f16_kernel<<<(FFd * Ed + 255) / 256, 256, 0, stream>>>(w2, w2t, FFd, Ed);

  // ---- phase 1: Q/K/V projections (WMMA) ----
  dim3 blk(256);
  dim3 gE(Ed / BN, Mtot / BM);
  gemm_f16_kernel<<<gE, blk, 0, stream>>>(xb, wqt, nullptr, Qf, nullptr, Mtot, Ed, Ed, 0);
  gemm_f16_kernel<<<gE, blk, 0, stream>>>(xb, wkt, nullptr, Kf, nullptr, Mtot, Ed, Ed, 0);
  gemm_f16_kernel<<<gE, blk, 0, stream>>>(xb, wvt, nullptr, Vf, nullptr, Mtot, Ed, Ed, 0);

  // ---- phase 2: rank-1 "attention" ----
  energy_kernel<<<(Mtot * Hh + 255) / 256, blk, 0, stream>>>(Qf, Kf, att);
  softmax_kernel<<<Nbat * Hh * Hh, blk, 0, stream>>>(att);
  asum_kernel<<<(Nbat * Hh * Sq + 255) / 256, blk, 0, stream>>>(att, Asum);
  vsum_kernel<<<(Mtot * Dd + 255) / 256, blk, 0, stream>>>(Vf, Vs);
  out2_kernel<<<(Nbat * Hh * Sq + 255) / 256, blk, 0, stream>>>(Asum, Vs, out2b);

  // ---- phase 3: O-projection + bias, residual, LN1 ----
  gemm_f16_kernel<<<gE, blk, 0, stream>>>(out2b, wot, bo, Of, nullptr, Mtot, Ed, Ed, 0);
  ln_kernel<<<Mtot, blk, 0, stream>>>(Of, x, g1, be1, x1, x1b);

  // ---- phase 4: FFN (relu fused into first GEMM, f16 out), LN2 -> d_out ----
  dim3 gF(FFd / BN, Mtot / BM);
  gemm_f16_kernel<<<gF, blk, 0, stream>>>(x1b, w1t, b1, nullptr, h1b, Mtot, FFd, Ed, 1);
  gemm_f16_kernel<<<gE, blk, 0, stream>>>(h1b, w2t, b2, ff, nullptr, Mtot, Ed, FFd, 0);
  ln_kernel<<<Mtot, blk, 0, stream>>>(ff, x1, g2, be2, (float*)d_out, nullptr);
}
